// HierarchicalMultiModalGNN_61761629716511
// MI455X (gfx1250) — matrix operand order repaired and checked by reading, not compiled, not run on hardware
//
#include <hip/hip_runtime.h>
#include <math.h>

#define EMB    128
#define HEADS  4
#define DH     32
#define NGRAPH 8

typedef __attribute__((ext_vector_type(16))) __bf16 v16bf;
typedef __attribute__((ext_vector_type(8)))  float  v8f;

// ---------------------------------------------------------------- utilities

__device__ __forceinline__ void atomicMaxF(float* addr, float v) {
  // works with buffers initialized to -inf (0xFF800000)
  if (v >= 0.0f) atomicMax((int*)addr, __float_as_int(v));
  else           atomicMin((unsigned int*)addr, (unsigned int)__float_as_int(v));
}

__global__ void fill_f32(float* __restrict__ p, float v, int n) {
  int i = blockIdx.x * blockDim.x + threadIdx.x;
  if (i < n) p[i] = v;
}

// --------------------------------------- weight pre-swizzle (fp32 -> bf16)
// Swizzled layout indexed by [kt = k/32][ncol][half][e(16)], 32 contiguous
// bf16 per (kt,ncol) pair -> each lane's B fragment is one 32-byte run.
__global__ void swizzle_w_bf16(const float* __restrict__ W, __bf16* __restrict__ Wsw,
                               int K, int NOUT) {
  int i = blockIdx.x * blockDim.x + threadIdx.x;
  if (i >= K * NOUT) return;
  int idx = i;
  const int e    = idx & 15;  idx >>= 4;
  const int half = idx & 1;   idx >>= 1;
  const int ncol = idx % NOUT;
  const int kt   = idx / NOUT;
  const int k    = kt * 32 + (e < 8 ? half * 8 + e : 16 + half * 8 + (e - 8));   // B layout: k = half*8 + e (e<8), 16 + half*8 + (e-8) (e>=8)
  Wsw[i] = (__bf16)W[(size_t)k * NOUT + ncol];
}

// ------------------------------------------------- WMMA GEMM (bf16 -> f32)
// Y[N x NOUT] = op(X[N x K] @ W[K x NOUT] + bias),  op = optional ReLU.
// Block = 256 (8 waves); each wave computes a 16 x NOUT strip: the A
// fragment is loaded once per K-step and reused across NOUT/16 accumulators.
// Fully unrolled -> dense v_wmma_f32_16x16x32_bf16 sequences.
template <int K, int NOUT>
__global__ void gemm_wmma(const float* __restrict__ X,
                          const __bf16* __restrict__ Wsw,
                          const float* __restrict__ B,
                          float* __restrict__ Y,
                          int N, int relu) {
  constexpr int NT = NOUT / 16;
  const int lane = threadIdx.x & 31;
  const int wv   = threadIdx.x >> 5;
  const int half = lane >> 4;
  const int l16  = lane & 15;
  const int tm   = blockIdx.x * 8 + wv;     // 16-row tile handled by this wave
  const int mrow = tm * 16 + l16;           // A-row this lane loads

  v8f acc[NT];
#pragma unroll
  for (int t = 0; t < NT; ++t) { v8f z = {}; acc[t] = z; }

#pragma unroll
  for (int kt = 0; kt < K / 32; ++kt) {
    v16bf a = {};
    // A 16x32 bf16 layout: elems 0..7 -> k = kt*32+half*8+j ; 8..15 -> +16
    if (mrow < N) {
      const float* xr = X + (size_t)mrow * K + kt * 32 + half * 8;
      const float4 x0 = *(const float4*)(xr + 0);
      const float4 x1 = *(const float4*)(xr + 4);
      const float4 x2 = *(const float4*)(xr + 16);
      const float4 x3 = *(const float4*)(xr + 20);
      a[0]=(__bf16)x0.x; a[1]=(__bf16)x0.y; a[2]=(__bf16)x0.z; a[3]=(__bf16)x0.w;
      a[4]=(__bf16)x1.x; a[5]=(__bf16)x1.y; a[6]=(__bf16)x1.z; a[7]=(__bf16)x1.w;
      a[8]=(__bf16)x2.x; a[9]=(__bf16)x2.y; a[10]=(__bf16)x2.z; a[11]=(__bf16)x2.w;
      a[12]=(__bf16)x3.x; a[13]=(__bf16)x3.y; a[14]=(__bf16)x3.z; a[15]=(__bf16)x3.w;
    }
#pragma unroll
    for (int t = 0; t < NT; ++t) {
      const int ncol = t * 16 + l16;
      const v16bf b = *(const v16bf*)(Wsw + ((size_t)kt * NOUT + ncol) * 32 + half * 16);
      acc[t] = __builtin_amdgcn_wmma_f32_16x16x32_bf16(
          /*neg_a=*/false, a, /*neg_b=*/false, b,
          /*c_mod=*/(short)0, acc[t], /*reuse_a=*/false, /*reuse_b=*/false);
    }
  }

#pragma unroll
  for (int t = 0; t < NT; ++t) {
    const int ncol = t * 16 + l16;
    const float bb = B[ncol];
#pragma unroll
    for (int r = 0; r < 8; ++r) {           // D: VGPR r -> M = r + 8*half
      const int m = tm * 16 + half * 8 + r;
      if (m < N) {
        float y = acc[t][r] + bb;
        if (relu) y = fmaxf(y, 0.0f);
        Y[(size_t)m * NOUT + ncol] = y;
      }
    }
  }
}

// -------------------------------------------- tiny dense layers (scalar)
__global__ void lin_small(const float* __restrict__ X, const float* __restrict__ W,
                          const float* __restrict__ B, float* __restrict__ Y,
                          int N, int K, int NOUT, int relu) {
  int i = blockIdx.x * blockDim.x + threadIdx.x;
  if (i >= N * NOUT) return;
  int n = i / NOUT, j = i % NOUT;
  float s = B[j];
  for (int k = 0; k < K; ++k) s = fmaf(X[(size_t)n * K + k], W[(size_t)k * NOUT + j], s);
  if (relu) s = fmaxf(s, 0.0f);
  Y[i] = s;
}

// ---------------------------------------- edge passes (TransformerConv)
// pass 1: raw attention logits + per-(dst,head) running max
__global__ void edge_alpha(const int* __restrict__ src, const int* __restrict__ dst,
                           const float* __restrict__ ea,
                           const float* __restrict__ Q, const float* __restrict__ Kf,
                           const float* __restrict__ eW,
                           float* __restrict__ alpha, float* __restrict__ amax, int E) {
  int i = blockIdx.x * blockDim.x + threadIdx.x;
  if (i >= E * HEADS) return;
  const int e = i >> 2, h = i & 3;
  const int s = src[e], d = dst[e];
  const float w = ea[e];
  const float* qp = Q  + (size_t)d * EMB + h * DH;
  const float* kp = Kf + (size_t)s * EMB + h * DH;
  const float* ep = eW + h * DH;
  float acc = 0.0f;
#pragma unroll 8
  for (int j = 0; j < DH; ++j) acc = fmaf(qp[j], fmaf(w, ep[j], kp[j]), acc);
  acc *= 0.17677669529663687f;             // 1/sqrt(DH)
  alpha[i] = acc;
  atomicMaxF(&amax[d * HEADS + h], acc);
}

// pass 2: exp(logit - max), accumulate softmax denominator
__global__ void edge_expsum(const int* __restrict__ dst, float* __restrict__ alpha,
                            const float* __restrict__ amax, float* __restrict__ denom,
                            int E) {
  int i = blockIdx.x * blockDim.x + threadIdx.x;
  if (i >= E * HEADS) return;
  const int e = i >> 2, h = i & 3;
  const int d = dst[e];
  const float w = __expf(alpha[i] - amax[d * HEADS + h]);
  alpha[i] = w;
  atomicAdd(&denom[d * HEADS + h], w);
}

// pass 3: weighted message scatter-add into agg (pre-seeded with skip)
__global__ void edge_agg(const int* __restrict__ src, const int* __restrict__ dst,
                         const float* __restrict__ ea,
                         const float* __restrict__ V, const float* __restrict__ eW,
                         const float* __restrict__ alpha, const float* __restrict__ denom,
                         float* __restrict__ agg, int E) {
  int i = blockIdx.x * blockDim.x + threadIdx.x;
  if (i >= E * EMB) return;
  const int e = i >> 7, c = i & 127, h = c >> 5;
  const int s = src[e], d = dst[e];
  const float w   = alpha[e * HEADS + h] / denom[d * HEADS + h];
  const float msg = fmaf(ea[e], eW[c], V[(size_t)s * EMB + c]) * w;
  atomicAdd(&agg[(size_t)d * EMB + c], msg);
}

// ------------------------------------------- fused LayerNorm + ReLU (wave/node)
__global__ void ln_relu(const float* __restrict__ X, const float* __restrict__ g,
                        const float* __restrict__ b, float* __restrict__ Y, int N) {
  const int lane = threadIdx.x & 31;
  const int node = blockIdx.x * (blockDim.x >> 5) + (threadIdx.x >> 5);
  if (node >= N) return;
  const float4 v = *(const float4*)(X + (size_t)node * EMB + lane * 4);
  float s = v.x + v.y + v.z + v.w;
#pragma unroll
  for (int m = 16; m; m >>= 1) s += __shfl_xor(s, m, 32);
  const float mu = s * (1.0f / EMB);
  const float d0 = v.x - mu, d1 = v.y - mu, d2 = v.z - mu, d3 = v.w - mu;
  float q = d0 * d0 + d1 * d1 + d2 * d2 + d3 * d3;
#pragma unroll
  for (int m = 16; m; m >>= 1) q += __shfl_xor(q, m, 32);
  const float rstd = rsqrtf(q * (1.0f / EMB) + 1e-5f);
  const float4 gg = *(const float4*)(g + lane * 4);
  const float4 bb = *(const float4*)(b + lane * 4);
  float4 o;
  o.x = fmaxf(fmaf(d0 * rstd, gg.x, bb.x), 0.0f);
  o.y = fmaxf(fmaf(d1 * rstd, gg.y, bb.y), 0.0f);
  o.z = fmaxf(fmaf(d2 * rstd, gg.z, bb.z), 0.0f);
  o.w = fmaxf(fmaf(d3 * rstd, gg.w, bb.w), 0.0f);
  *(float4*)(Y + (size_t)node * EMB + lane * 4) = o;
}

// ------------------------------------------- global attention pooling
__global__ void gate_scalar(const float* __restrict__ GH, const float* __restrict__ W2,
                            const float* __restrict__ B2, const int* __restrict__ batch,
                            float* __restrict__ gate, float* __restrict__ gmax,
                            int N, int K) {
  int n = blockIdx.x * blockDim.x + threadIdx.x;
  if (n >= N) return;
  float s = B2[0];
  for (int k = 0; k < K; ++k) s = fmaf(GH[(size_t)n * K + k], W2[k], s);
  gate[n] = s;
  atomicMaxF(&gmax[batch[n]], s);
}

__global__ void gate_exp(float* __restrict__ gate, const int* __restrict__ batch,
                         const float* __restrict__ gmax, float* __restrict__ gsum, int N) {
  int n = blockIdx.x * blockDim.x + threadIdx.x;
  if (n >= N) return;
  const int b = batch[n];
  const float w = __expf(gate[n] - gmax[b]);
  gate[n] = w;
  atomicAdd(&gsum[b], w);
}

__global__ void pool_scatter(const float* __restrict__ X, const float* __restrict__ gate,
                             const float* __restrict__ gsum, const int* __restrict__ batch,
                             float* __restrict__ hbuf, int N, int off) {
  int i = blockIdx.x * blockDim.x + threadIdx.x;
  if (i >= N * EMB) return;
  const int n = i >> 7, c = i & 127;
  const int g = batch[n];
  atomicAdd(&hbuf[(size_t)g * (2 * EMB) + off + c], X[i] * gate[n] / gsum[g]);
}

// ------------------------------------------------------------- host side

static inline void fillf(hipStream_t s, float* p, float v, size_t n) {
  fill_f32<<<dim3((unsigned)((n + 255) / 256)), dim3(256), 0, s>>>(p, v, (int)n);
}

static inline void swz(hipStream_t s, const float* W, __bf16* Wsw, int K, int NOUT) {
  const int n = K * NOUT;
  swizzle_w_bf16<<<dim3((n + 255) / 256), dim3(256), 0, s>>>(W, Wsw, K, NOUT);
}

template <int K, int NOUT>
static inline void launch_gemm(hipStream_t s, const float* X, const __bf16* Wsw,
                               const float* B, float* Y, int N, int relu) {
  gemm_wmma<K, NOUT><<<dim3((N + 127) / 128), dim3(256), 0, s>>>(X, Wsw, B, Y, N, relu);
}

static inline void run_edges(hipStream_t s, const int* src, const int* dst,
                             const float* ea, const float* Q, const float* Kf,
                             const float* V, const float* eW,
                             float* alpha, float* amax, float* denom,
                             float* agg, int E, int Ndst) {
  fillf(s, amax, -INFINITY, (size_t)Ndst * HEADS);
  fillf(s, denom, 0.0f, (size_t)Ndst * HEADS);
  const int t1 = E * HEADS;
  edge_alpha<<<dim3((t1 + 255) / 256), dim3(256), 0, s>>>(src, dst, ea, Q, Kf, eW,
                                                          alpha, amax, E);
  edge_expsum<<<dim3((t1 + 255) / 256), dim3(256), 0, s>>>(dst, alpha, amax, denom, E);
  const int t2 = E * EMB;
  edge_agg<<<dim3((t2 + 255) / 256), dim3(256), 0, s>>>(src, dst, ea, V, eW,
                                                        alpha, denom, agg, E);
}

extern "C" void kernel_launch(void* const* d_in, const int* in_sizes, int n_in,
                              void* d_out, int out_size, void* d_ws, size_t ws_size,
                              hipStream_t stream) {
  (void)n_in; (void)out_size; (void)ws_size;
#define FIN(i) ((const float*)d_in[i])
#define IIN(i) ((const int*)d_in[i])
  const int NC  = in_sizes[12];
  const int NG  = in_sizes[13];
  const int Ecc = in_sizes[3], Egg = in_sizes[5];
  const int Ecg = in_sizes[6], Egc = in_sizes[9];
  const int* src_cc = IIN(2);          const int* dst_cc = IIN(2) + Ecc;
  const int* src_gg = IIN(4);          const int* dst_gg = IIN(4) + Egg;
  const int* src_cg = IIN(6);          const int* dst_cg = IIN(7);
  const int* src_gc = IIN(9);          const int* dst_gc = IIN(10);
  const float *ea_cc = FIN(3), *ea_gg = FIN(5), *ea_cg = FIN(8), *ea_gc = FIN(11);
  const int* batch_c = IIN(12);        const int* batch_g = IIN(13);
  // params: tconv blocks at 14(cc),23(gg),32(cg),41(gc); layout:
  // +0 qW +1 qb +2 kW +3 kb +4 vW +5 vb +6 eW +7 skipW +8 skipb
  const int CC = 14, GG = 23, CG = 32, GC = 41;
  // 50/51 ln_c1, 52/53 ln_g1, 54/55 ln_c2, 56/57 ln_g2
  // 58..61 gate_cell, 62..65 gate_gene, 66..69 readout

  // ---------------- workspace layout (floats) ----------------
  float* w = (float*)d_ws;
  size_t o = 0;
  auto A = [&](size_t nf) { float* p = w + o; o += nf; return p; };
  auto Abf = [&](size_t nbf) { __bf16* p = (__bf16*)(w + o); o += (nbf + 7) / 8 * 4; return p; };
  float* cQ = A((size_t)NC * EMB);   // stage2: K_cg
  float* cK = A((size_t)NC * EMB);   // stage2: V_cg
  float* cV = A((size_t)NC * EMB);   // stage2: Q_gc
  float* cS = A((size_t)NC * EMB);   // agg for cell (skip-seeded)
  float* gQ = A((size_t)NG * EMB);   // stage2: Q_cg
  float* gK = A((size_t)NG * EMB);   // stage2: K_gc
  float* gV = A((size_t)NG * EMB);   // stage2: V_gc
  float* gS = A((size_t)NG * EMB);   // agg for gene (skip-seeded)
  float* c1 = A((size_t)NC * EMB);
  float* g1 = A((size_t)NG * EMB);
  size_t Emax = (size_t)(Ecc > Egg ? Ecc : Egg);
  if ((size_t)Ecg > Emax) Emax = Ecg;
  if ((size_t)Egc > Emax) Emax = Egc;
  size_t ascratch = Emax * HEADS;
  if ((size_t)NC * (EMB / 2) > ascratch) ascratch = (size_t)NC * (EMB / 2);
  float* alpha = A(ascratch);        // later reused as gate-MLP hidden
  float* amax  = A((size_t)NC * HEADS);  // later reused as gate scalar
  float* denom = A((size_t)NC * HEADS);
  float* gmax  = A(NGRAPH);
  float* gsum  = A(NGRAPH);
  float* hbuf  = A((size_t)NGRAPH * 2 * EMB);
  float* r1    = A((size_t)NGRAPH * 2 * EMB);
  // swizzled bf16 weights (WMMA B-fragment layout)
  __bf16* wsw[12];                       // cc q,k,v,skip | gg ... | cg ... (below)
  const int CIN = 64, GIN = 128;
  for (int t = 0; t < 4; ++t) wsw[t]     = Abf((size_t)CIN * EMB);   // cc
  for (int t = 0; t < 4; ++t) wsw[4 + t] = Abf((size_t)GIN * EMB);   // gg
  __bf16* wcg[4]; for (int t = 0; t < 4; ++t) wcg[t] = Abf((size_t)EMB * EMB);
  __bf16* wgc[4]; for (int t = 0; t < 4; ++t) wgc[t] = Abf((size_t)EMB * EMB);
  __bf16* wgatec = Abf((size_t)EMB * (EMB / 2));
  __bf16* wgateg = Abf((size_t)EMB * (EMB / 2));
  __bf16* wro    = Abf((size_t)(2 * EMB) * (2 * EMB));

  // ---------------- weight pre-swizzle (runs every call; deterministic) ----
  const int WOFF[4] = {0, 2, 4, 7};      // q, k, v, skip within a tconv block
  for (int t = 0; t < 4; ++t) swz(stream, FIN(CC + WOFF[t]), wsw[t],     CIN, EMB);
  for (int t = 0; t < 4; ++t) swz(stream, FIN(GG + WOFF[t]), wsw[4 + t], GIN, EMB);
  for (int t = 0; t < 4; ++t) swz(stream, FIN(CG + WOFF[t]), wcg[t],     EMB, EMB);
  for (int t = 0; t < 4; ++t) swz(stream, FIN(GC + WOFF[t]), wgc[t],     EMB, EMB);
  swz(stream, FIN(58), wgatec, EMB, EMB / 2);
  swz(stream, FIN(62), wgateg, EMB, EMB / 2);
  swz(stream, FIN(66), wro, 2 * EMB, 2 * EMB);

  // ---------------- stage 1: intra-modal TransformerConvs ----------------
  const float* xc = FIN(0);
  const float* xg = FIN(1);
  launch_gemm<64, 128>(stream, xc, wsw[0], FIN(CC + 1), cQ, NC, 0);
  launch_gemm<64, 128>(stream, xc, wsw[1], FIN(CC + 3), cK, NC, 0);
  launch_gemm<64, 128>(stream, xc, wsw[2], FIN(CC + 5), cV, NC, 0);
  launch_gemm<64, 128>(stream, xc, wsw[3], FIN(CC + 8), cS, NC, 0);  // skip seed
  launch_gemm<128, 128>(stream, xg, wsw[4], FIN(GG + 1), gQ, NG, 0);
  launch_gemm<128, 128>(stream, xg, wsw[5], FIN(GG + 3), gK, NG, 0);
  launch_gemm<128, 128>(stream, xg, wsw[6], FIN(GG + 5), gV, NG, 0);
  launch_gemm<128, 128>(stream, xg, wsw[7], FIN(GG + 8), gS, NG, 0);

  run_edges(stream, src_cc, dst_cc, ea_cc, cQ, cK, cV, FIN(CC + 6),
            alpha, amax, denom, cS, Ecc, NC);
  run_edges(stream, src_gg, dst_gg, ea_gg, gQ, gK, gV, FIN(GG + 6),
            alpha, amax, denom, gS, Egg, NG);

  ln_relu<<<dim3((NC + 7) / 8), dim3(256), 0, stream>>>(cS, FIN(50), FIN(51), c1, NC);
  ln_relu<<<dim3((NG + 7) / 8), dim3(256), 0, stream>>>(gS, FIN(52), FIN(53), g1, NG);

  // ---------------- stage 2: inter-modal TransformerConvs ----------------
  // cg: src = cells(c1), dst = genes(g1)
  launch_gemm<128, 128>(stream, g1, wcg[0], FIN(CG + 1), gQ, NG, 0);
  launch_gemm<128, 128>(stream, c1, wcg[1], FIN(CG + 3), cQ, NC, 0);  // K_cg
  launch_gemm<128, 128>(stream, c1, wcg[2], FIN(CG + 5), cK, NC, 0);  // V_cg
  launch_gemm<128, 128>(stream, g1, wcg[3], FIN(CG + 8), gS, NG, 0);  // skip seed
  // gc: src = genes(g1), dst = cells(c1)
  launch_gemm<128, 128>(stream, c1, wgc[0], FIN(GC + 1), cV, NC, 0);  // Q_gc
  launch_gemm<128, 128>(stream, g1, wgc[1], FIN(GC + 3), gK, NG, 0);
  launch_gemm<128, 128>(stream, g1, wgc[2], FIN(GC + 5), gV, NG, 0);
  launch_gemm<128, 128>(stream, c1, wgc[3], FIN(GC + 8), cS, NC, 0);

  run_edges(stream, src_cg, dst_cg, ea_cg, gQ, cQ, cK, FIN(CG + 6),
            alpha, amax, denom, gS, Ecg, NG);
  run_edges(stream, src_gc, dst_gc, ea_gc, cV, gK, gV, FIN(GC + 6),
            alpha, amax, denom, cS, Egc, NC);

  ln_relu<<<dim3((NC + 7) / 8), dim3(256), 0, stream>>>(cS, FIN(54), FIN(55), c1, NC);
  ln_relu<<<dim3((NG + 7) / 8), dim3(256), 0, stream>>>(gS, FIN(56), FIN(57), g1, NG);

  // ---------------- global attention pooling ----------------
  float* gh   = alpha;   // N x 64 hidden (reuse)
  float* gate = amax;    // per-node scalar (reuse)
  fillf(stream, hbuf, 0.0f, (size_t)NGRAPH * 2 * EMB);

  // cells -> hbuf[:, 0:128]
  fillf(stream, gmax, -INFINITY, NGRAPH);
  fillf(stream, gsum, 0.0f, NGRAPH);
  launch_gemm<128, 64>(stream, c1, wgatec, FIN(59), gh, NC, 1);
  gate_scalar<<<dim3((NC + 255) / 256), dim3(256), 0, stream>>>(
      gh, FIN(60), FIN(61), batch_c, gate, gmax, NC, EMB / 2);
  gate_exp<<<dim3((NC + 255) / 256), dim3(256), 0, stream>>>(gate, batch_c, gmax, gsum, NC);
  pool_scatter<<<dim3((NC * EMB + 255) / 256), dim3(256), 0, stream>>>(
      c1, gate, gsum, batch_c, hbuf, NC, 0);

  // genes -> hbuf[:, 128:256]
  fillf(stream, gmax, -INFINITY, NGRAPH);
  fillf(stream, gsum, 0.0f, NGRAPH);
  launch_gemm<128, 64>(stream, g1, wgateg, FIN(63), gh, NG, 1);
  gate_scalar<<<dim3((NG + 255) / 256), dim3(256), 0, stream>>>(
      gh, FIN(64), FIN(65), batch_g, gate, gmax, NG, EMB / 2);
  gate_exp<<<dim3((NG + 255) / 256), dim3(256), 0, stream>>>(gate, batch_g, gmax, gsum, NG);
  pool_scatter<<<dim3((NG * EMB + 255) / 256), dim3(256), 0, stream>>>(
      g1, gate, gsum, batch_g, hbuf, NG, EMB);

  // ---------------- readout ----------------
  launch_gemm<256, 256>(stream, hbuf, wro, FIN(67), r1, NGRAPH, 1);
  lin_small<<<dim3(1), dim3(64), 0, stream>>>(r1, FIN(68), FIN(69), (float*)d_out,
                                              NGRAPH, 2 * EMB, 4, 0);
#undef FIN
#undef IIN
}
